// DualGraphGAT_48412871360536
// MI455X (gfx1250) — matrix operand order repaired and checked by reading, not compiled, not run on hardware
//
#include <hip/hip_runtime.h>
#include <cstdint>
#include <cstddef>

// ---------------- problem constants (from reference) ----------------
#define BB 128
#define GG 16384
#define NCC 32
#define HDD 512
#define NHEADS 2
#define EGG 262144
#define EPP 2048

typedef unsigned short u16t;

// ---------------- WMMA types ----------------
typedef __bf16 bf16_t;
typedef __attribute__((ext_vector_type(16))) bf16_t v16bf;
typedef __attribute__((ext_vector_type(8)))  float  v8f;

union FragBF { v16bf v; unsigned u[8]; };

__device__ __forceinline__ u16t f2bf_rne(float f) {
    unsigned u = __float_as_uint(f);
    return (u16t)((u + 0x7FFFu + ((u >> 16) & 1u)) >> 16);
}

__device__ __forceinline__ unsigned encF(float f) {
    unsigned u = __float_as_uint(f);
    return (u & 0x80000000u) ? ~u : (u | 0x80000000u);
}
__device__ __forceinline__ float decF(unsigned u) {
    u = (u & 0x80000000u) ? (u & 0x7FFFFFFFu) : ~u;
    return __uint_as_float(u);
}

__device__ __forceinline__ float waveRed(float x) {
    #pragma unroll
    for (int o = 16; o; o >>= 1) x += __shfl_xor(x, o, 32);
    return x;
}

// CDNA5 async global->LDS copy (GVS mode, ASYNCcnt-tracked), 16B per lane.
__device__ __forceinline__ void async_b128(unsigned lds_off, unsigned goff,
                                           unsigned long long base) {
    asm volatile("global_load_async_to_lds_b128 %0, %1, %2 offset:0"
                 :: "v"(lds_off), "v"(goff), "s"(base) : "memory");
}
__device__ __forceinline__ void wait_async3() {
    asm volatile("s_wait_asynccnt 0x3" ::: "memory");
}
__device__ __forceinline__ void wait_async0() {
    asm volatile("s_wait_asynccnt 0x0" ::: "memory");
}

// ---------------- GEMM: C[M,N] = act( A[M,K] @ W[K,N] + bias ) ----------------
// A: bf16 [M][K]; Wt: bf16 [N][K] (pre-transposed). Tiles stream into LDS via
// double-buffered async global->LDS b128 (3 instr/wave/stage); fragments are
// plain ds_load_b32 k-pair gathers; 4x v_wmma_f32_16x16x32_bf16 per K-step.
// Block tile 128(M) x 64(N), 8 waves (4x2), each wave 32x32.
// LDS rows stride 80B (16B-aligned for async, conflict-free 20-dword stride).
__global__ __launch_bounds__(256)
void gemm_wmma_bf16(const u16t* __restrict__ A, const u16t* __restrict__ Wt,
                    const float* __restrict__ bias,
                    float* __restrict__ Cf, u16t* __restrict__ Cbf,
                    int M, int N, int K, int ldc, int colofs, int act) {
    __shared__ __align__(16) u16t sA[2][128][40];
    __shared__ __align__(16) u16t sB[2][64][40];

    const int tid  = threadIdx.x;
    const int wave = tid >> 5;
    const int lane = tid & 31;
    const int wm = wave & 3;     // 4 row-groups of 32
    const int wn = wave >> 2;    // 2 col-groups of 32
    const int m0 = blockIdx.y * 128;
    const int n0 = blockIdx.x * 64;

    const bool fast = (m0 + 128 <= M) && (n0 + 64 <= N);

    const unsigned aLds0 = (unsigned)(size_t)&sA[0][0][0];
    const unsigned bLds0 = (unsigned)(size_t)&sB[0][0][0];
    const unsigned long long aBase = (unsigned long long)(size_t)A;
    const unsigned long long wBase = (unsigned long long)(size_t)Wt;

    // one stage of async copies: per thread 2 A-chunks + 1 B-chunk (16B each)
    auto issue_stage = [&](int buf, int k0) {
        unsigned aL = aLds0 + (unsigned)buf * (128u * 40u * 2u);
        unsigned bL = bLds0 + (unsigned)buf * (64u * 40u * 2u);
        #pragma unroll
        for (int j = 0; j < 2; ++j) {                 // A: 512 16B-chunks
            unsigned c = (unsigned)tid + 256u * j;
            unsigned m = c >> 2, kc = c & 3;
            unsigned goff = (unsigned)(((size_t)(m0 + m) * K + k0) * 2) + (kc << 4);
            async_b128(aL + m * 80u + (kc << 4), goff, aBase);
        }
        {                                             // B: 256 16B-chunks
            unsigned n = (unsigned)tid >> 2, kc = (unsigned)tid & 3;
            unsigned goff = (unsigned)(((size_t)(n0 + n) * K + k0) * 2) + (kc << 4);
            async_b128(bL + n * 80u + (kc << 4), goff, wBase);
        }
    };

    v8f acc[2][2] = {};
    int cur = 0;

    if (fast) issue_stage(0, 0);

    for (int k0 = 0; k0 < K; k0 += 32) {
        if (fast) {
            if (k0 + 32 < K) { issue_stage(cur ^ 1, k0 + 32); wait_async3(); }
            else             { wait_async0(); }
        } else {
            // synchronous guarded fallback (partial-N tiles)
            for (int i = tid; i < 128 * 32; i += 256) {
                int m = i >> 5, k = i & 31;
                sA[0][m][k] = (m0 + m < M) ? A[(size_t)(m0 + m) * K + (k0 + k)] : (u16t)0;
            }
            for (int i = tid; i < 64 * 32; i += 256) {
                int n = i >> 5, k = i & 31;
                sB[0][n][k] = (n0 + n < N) ? Wt[(size_t)(n0 + n) * K + (k0 + k)] : (u16t)0;
            }
        }
        __syncthreads();

        // ---- fragments per documented wave32 VGPR layouts ----
        const int rsub  = lane & 15;
        const int khalf = lane >> 4;   // A: 0 -> K{0-7,16-23}, 1 -> K{8-15,24-31}
        FragBF a[2], b[2];
        #pragma unroll
        for (int t = 0; t < 2; ++t) {
            const u16t* pa = &sA[cur][wm * 32 + t * 16 + rsub][0];
            #pragma unroll
            for (int v = 0; v < 4; ++v)
                a[t].u[v] = *(const unsigned*)(pa + khalf * 8 + 2 * v);
            #pragma unroll
            for (int v = 0; v < 4; ++v)
                a[t].u[4 + v] = *(const unsigned*)(pa + 16 + khalf * 8 + 2 * v);
            const u16t* pb = &sB[cur][wn * 32 + t * 16 + rsub][0];
            #pragma unroll
            for (int v = 0; v < 8; ++v)
                b[t].u[v] = *(const unsigned*)(pb + (v >> 2) * 16 + khalf * 8 + 2 * (v & 3));
        }
        #pragma unroll
        for (int i = 0; i < 2; ++i)
            #pragma unroll
            for (int j = 0; j < 2; ++j)
                acc[i][j] = __builtin_amdgcn_wmma_f32_16x16x32_bf16(
                    false, a[i].v, false, b[j].v, (short)0, acc[i][j], false, false);
        __syncthreads();
        if (fast) cur ^= 1;
    }

    // ---- writeback: 16x16 f32 C layout: VGPR v -> M = v + (lane>=16 ? 8:0), N = lane&15 ----
    #pragma unroll
    for (int i = 0; i < 2; ++i) {
        #pragma unroll
        for (int j = 0; j < 2; ++j) {
            int col = n0 + wn * 32 + j * 16 + (lane & 15);
            if (col >= N) continue;
            float bv = bias ? bias[col] : 0.f;
            int rbase = m0 + wm * 32 + i * 16 + ((lane >> 4) * 8);
            #pragma unroll
            for (int v = 0; v < 8; ++v) {
                int row = rbase + v;
                if (row < M) {
                    float x = acc[i][j][v] + bv;
                    if (act == 1) x = fmaxf(x, 0.f);
                    if (Cf)  Cf [(size_t)row * ldc + colofs + col] = x;
                    if (Cbf) Cbf[(size_t)row * ldc + colofs + col] = f2bf_rne(x);
                }
            }
        }
    }
}

// ---------------- conversions ----------------
// centrality weighting; emits bf16 xg [B][G] and bf16 xg^T [G][B]
__global__ void k_xg(const float* __restrict__ xg_in, const float* __restrict__ cl,
                     const float* __restrict__ ev, const float* __restrict__ bt,
                     const float* __restrict__ al, const float* __restrict__ be,
                     const float* __restrict__ ga,
                     u16t* __restrict__ xg_bf, u16t* __restrict__ xgT_bf) {
    int idx = blockIdx.x * 256 + threadIdx.x;
    if (idx < BB * GG) {
        int b = idx >> 14;           // / GG
        int g = idx & (GG - 1);
        float c = al[0] * cl[idx] + be[0] * ev[idx] + ga[0] * bt[idx];
        u16t v = f2bf_rne(xg_in[idx] * c);
        xg_bf[idx] = v;
        xgT_bf[(size_t)g * BB + b] = v;
    }
}

__global__ void k_cvt(const float* __restrict__ in, u16t* __restrict__ out, int n) {
    int i = blockIdx.x * 256 + threadIdx.x;
    if (i < n) out[i] = f2bf_rne(in[i]);
}

// fp32 [K][N] -> bf16 [N][K], LDS-tiled 32x32 transpose
__global__ __launch_bounds__(256)
void k_cvtT(const float* __restrict__ in, u16t* __restrict__ out, int K, int N) {
    __shared__ float tile[32][33];
    int n0 = blockIdx.x * 32, k0 = blockIdx.y * 32;
    int tid = threadIdx.x;
    for (int idx = tid; idx < 1024; idx += 256) {
        int i = idx >> 5, j = idx & 31;
        tile[i][j] = (k0 + i < K && n0 + j < N) ? in[(size_t)(k0 + i) * N + (n0 + j)] : 0.f;
    }
    __syncthreads();
    for (int idx = tid; idx < 1024; idx += 256) {
        int i = idx >> 5, j = idx & 31;
        if (n0 + i < N && k0 + j < K)
            out[(size_t)(n0 + i) * K + (k0 + j)] = f2bf_rne(tile[j][i]);
    }
}

__global__ void k_fillf(float* p, float v, int n) {
    int i = blockIdx.x * 256 + threadIdx.x;
    if (i < n) p[i] = v;
}
__global__ void k_fillu(unsigned* p, unsigned v, int n) {
    int i = blockIdx.x * 256 + threadIdx.x;
    if (i < n) p[i] = v;
}

// ---------------- GATv2 edge pipeline ----------------
__global__ void k_edge_scores(const float* __restrict__ xl, const float* __restrict__ xr,
                              const float* __restrict__ att,
                              const int* __restrict__ src, const int* __restrict__ dst,
                              int E, int N, int C,
                              float* __restrict__ esc, unsigned* __restrict__ emax) {
    int gw = (blockIdx.x * blockDim.x + threadIdx.x) >> 5;
    int nw = (gridDim.x * blockDim.x) >> 5;
    int lane = threadIdx.x & 31;
    int total = E + N;
    int HC = NHEADS * C;
    for (int e = gw; e < total; e += nw) {
        int s = (e < E) ? src[e] : (e - E);
        int d = (e < E) ? dst[e] : (e - E);
        const float* ps = xl + (size_t)s * HC;
        const float* pd = xr + (size_t)d * HC;
        #pragma unroll
        for (int h = 0; h < NHEADS; ++h) {
            float a = 0.f;
            for (int c = lane; c < C; c += 32) {
                float x = ps[h * C + c] + pd[h * C + c];
                x = (x > 0.f) ? x : 0.2f * x;           // GATv2 leaky slope 0.2
                a += x * att[h * C + c];
            }
            a = waveRed(a);
            if (lane == 0) {
                esc[(size_t)e * NHEADS + h] = a;
                atomicMax(&emax[d * NHEADS + h], encF(a));
            }
        }
    }
}

__global__ void k_edge_soft(const float* __restrict__ esc, const int* __restrict__ dst,
                            int E, int N, const unsigned* __restrict__ emax,
                            float* __restrict__ ew, float* __restrict__ den) {
    int idx = blockIdx.x * 256 + threadIdx.x;
    int total = (E + N) * NHEADS;
    if (idx < total) {
        int e = idx / NHEADS, h = idx - e * NHEADS;
        int d = (e < E) ? dst[e] : (e - E);
        float w = __expf(esc[idx] - decF(emax[d * NHEADS + h]));
        ew[idx] = w;
        atomicAdd(&den[d * NHEADS + h], w);
    }
}

__global__ void k_edge_scatter(const float* __restrict__ xl,
                               const int* __restrict__ src, const int* __restrict__ dst,
                               int E, int N, int C,
                               const float* __restrict__ ew, const float* __restrict__ den,
                               float* __restrict__ accum) {
    int gw = (blockIdx.x * blockDim.x + threadIdx.x) >> 5;
    int nw = (gridDim.x * blockDim.x) >> 5;
    int lane = threadIdx.x & 31;
    int total = E + N;
    int HC = NHEADS * C;
    for (int e = gw; e < total; e += nw) {
        int s = (e < E) ? src[e] : (e - E);
        int d = (e < E) ? dst[e] : (e - E);
        #pragma unroll
        for (int h = 0; h < NHEADS; ++h) {
            float alpha = ew[(size_t)e * NHEADS + h] / den[d * NHEADS + h];
            const float* ps = xl + (size_t)s * HC + h * C;
            float* pa = accum + (size_t)d * HC + h * C;
            for (int c = lane; c < C; c += 32)
                atomicAdd(&pa[c], alpha * ps[c]);
        }
    }
}

// Finalize (concat or head-mean) + bias + LayerNorm + activation (+ residual).
// Optional fp32 output, optional bf16 mirror (optionally transposed [D][N]).
__global__ void k_gat_post(const float* __restrict__ accum, const float* __restrict__ bias,
                           const float* __restrict__ res,
                           float* __restrict__ out, u16t* __restrict__ outbf, int transbf,
                           int N, int Hn, int C, int concat, int act) {
    int gw = (blockIdx.x * blockDim.x + threadIdx.x) >> 5;
    int nw = (gridDim.x * blockDim.x) >> 5;
    int lane = threadIdx.x & 31;
    int D = concat ? Hn * C : C;
    float invH = 1.f / (float)Hn;
    for (int n = gw; n < N; n += nw) {
        const float* pa = accum + (size_t)n * Hn * C;
        float sum = 0.f, sq = 0.f;
        for (int d = lane; d < D; d += 32) {
            float x;
            if (concat) x = pa[d];
            else { x = 0.f; for (int h = 0; h < Hn; ++h) x += pa[h * C + d]; x *= invH; }
            if (bias) x += bias[d];
            sum += x; sq += x * x;
        }
        sum = waveRed(sum); sq = waveRed(sq);
        float m  = sum / (float)D;
        float var = sq / (float)D - m * m;
        float rs = rsqrtf(var + 1e-5f);
        for (int d = lane; d < D; d += 32) {
            float x;
            if (concat) x = pa[d];
            else { x = 0.f; for (int h = 0; h < Hn; ++h) x += pa[h * C + d]; x *= invH; }
            if (bias) x += bias[d];
            x = (x - m) * rs;
            if (act == 0) x = (x > 0.f) ? x : 0.01f * x;   // leaky_relu 0.01
            else          x = fmaxf(x, 0.f);               // relu
            if (res) x += res[(size_t)n * D + d];
            if (out) out[(size_t)n * D + d] = x;
            if (outbf) {
                u16t bf = f2bf_rne(x);
                if (transbf) outbf[(size_t)d * N + n] = bf;
                else         outbf[(size_t)n * D + d] = bf;
            }
        }
    }
}

// out[n] = dot(X[n,:], w) + b   (N=1 output heads)
__global__ void k_dot(const float* __restrict__ X, const float* __restrict__ w,
                      const float* __restrict__ b, float* __restrict__ out,
                      int N, int K) {
    int gw = (blockIdx.x * blockDim.x + threadIdx.x) >> 5;
    int nw = (gridDim.x * blockDim.x) >> 5;
    int lane = threadIdx.x & 31;
    for (int n = gw; n < N; n += nw) {
        float s = 0.f;
        for (int k = lane; k < K; k += 32) s += X[(size_t)n * K + k] * w[k];
        s = waveRed(s);
        if (lane == 0) out[n] = s + b[0];
    }
}

// ---------------- host orchestration ----------------
extern "C" void kernel_launch(void* const* d_in, const int* in_sizes, int n_in,
                              void* d_out, int out_size, void* d_ws, size_t ws_size,
                              hipStream_t stream) {
    (void)in_sizes; (void)n_in; (void)out_size; (void)ws_size;
    auto F = [&](int i) { return (const float*)d_in[i]; };
    auto cdiv = [](int a, int b) { return (a + b - 1) / b; };

    // pytree leaf order: dict keys sorted alphabetically, tuples in order
    const float* x_genes = F(0);
    const int*   gsrc = (const int*)d_in[1];           const int* gdst = gsrc + EGG;
    const int*   psrc = (const int*)d_in[2];           const int* pdst = psrc + EPP;
    const float* clinical = F(3);
    const float* cl = F(4); const float* ev = F(5); const float* bt = F(6);
    const float* alpha = F(7); const float* beta = F(8);
    const float* clinW = F(9);  const float* clinB = F(10);
    const float* fc1W = F(11);  const float* fc1b = F(12);
    const float* fc2W = F(13);  const float* fc2b = F(14);
    const float* fc3W = F(15);  const float* fc3b = F(16);
    const float* g2pW = F(17);  const float* g2pb = F(18);
    const float* fgW  = F(19);  const float* fgb  = F(20);
    const float* fpW  = F(21);  const float* fpb  = F(22);
    const float* gamma = F(23);
    // gg1(24..29) gg2(30..35) gg3(36..41) gp1(42..47) gp2(48..53) gp3(54..59)
    // mha(60..67, dead code: contributes exactly 0), risk1..3(68..73)
    const float* r1W = F(68); const float* r1b = F(69);
    const float* r2W = F(70); const float* r2b = F(71);
    const float* r3W = F(72); const float* r3b = F(73);

    // ----- workspace arena -----
    char* wsb = (char*)d_ws;
    size_t off = 0;
    auto alloc = [&](size_t bytes) { void* p = wsb + off; off = (off + bytes + 255) & ~(size_t)255; return p; };
    u16t*     xg_bf  = (u16t*)alloc((size_t)BB * GG * 2);
    u16t*     xgT_bf = (u16t*)alloc((size_t)GG * BB * 2);
    float*    gxl  = (float*)alloc((size_t)GG * 256 * 4);
    float*    gxr  = (float*)alloc((size_t)GG * 256 * 4);
    float*    gacc = (float*)alloc((size_t)GG * 256 * 4);
    u16t*     gh1_bf  = (u16t*)alloc((size_t)GG * 256 * 2);
    float*    gh2     = (float*)alloc((size_t)GG * 128 * 4);
    u16t*     gh2_bf  = (u16t*)alloc((size_t)GG * 128 * 2);
    u16t*     gh3T_bf = (u16t*)alloc((size_t)BB * GG * 2);
    float*    gesc = (float*)alloc((size_t)(EGG + GG) * NHEADS * 4);
    float*    gewt = (float*)alloc((size_t)(EGG + GG) * NHEADS * 4);
    unsigned* gemx = (unsigned*)alloc((size_t)GG * NHEADS * 4);
    float*    gden = (float*)alloc((size_t)GG * NHEADS * 4);
    u16t*     pq_bf = (u16t*)alloc((size_t)BB * 512 * 2);
    float*    pxl  = (float*)alloc((size_t)BB * 1024 * 4);
    float*    pxr  = (float*)alloc((size_t)BB * 1024 * 4);
    float*    pacc = (float*)alloc((size_t)BB * 1024 * 4);
    u16t*     ph1_bf = (u16t*)alloc((size_t)BB * 1024 * 2);
    float*    ph2    = (float*)alloc((size_t)BB * 512 * 4);
    u16t*     ph2_bf = (u16t*)alloc((size_t)BB * 512 * 2);
    u16t*     ph3_bf = (u16t*)alloc((size_t)BB * 512 * 2);
    float*    pesc = (float*)alloc((size_t)(EPP + BB) * NHEADS * 4);
    float*    pewt = (float*)alloc((size_t)(EPP + BB) * NHEADS * 4);
    unsigned* pemx = (unsigned*)alloc((size_t)BB * NHEADS * 4);
    float*    pden = (float*)alloc((size_t)BB * NHEADS * 4);
    u16t*     comb_bf = (u16t*)alloc((size_t)BB * 1056 * 2);
    u16t*     clin_bf = (u16t*)alloc((size_t)BB * NCC * 2);
    u16t*     r1_bf = (u16t*)alloc((size_t)BB * 512 * 2);
    float*    r2   = (float*)alloc((size_t)BB * 64 * 4);
    float*    t1   = (float*)alloc((size_t)BB * 256 * 4);
    u16t*     t1_bf = (u16t*)alloc((size_t)BB * 256 * 2);
    float*    t2   = (float*)alloc((size_t)BB * 32 * 4);
    u16t*     wbufL = (u16t*)alloc((size_t)HDD * GG * 2);   // largest W^T: 512x16384
    u16t*     wbufR = (u16t*)alloc((size_t)HDD * GG * 2);

    auto gemm = [&](const u16t* A, const u16t* Wt, const float* bias,
                    float* Cf, u16t* Cbf,
                    int M, int N, int K, int ldc, int colofs, int act) {
        dim3 g(cdiv(N, 64), cdiv(M, 128));
        gemm_wmma_bf16<<<g, 256, 0, stream>>>(A, Wt, bias, Cf, Cbf, M, N, K, ldc, colofs, act);
    };
    auto cvtT = [&](const float* Win, u16t* outT, int Kd, int Nd) {
        dim3 g(cdiv(Nd, 32), cdiv(Kd, 32));
        k_cvtT<<<g, 256, 0, stream>>>(Win, outT, Kd, Nd);
    };

    // one GATv2 layer: A_bf [Nn][Fin] -> out / outbf [Nn, concat?2C:C]
    auto gat_layer = [&](const u16t* A_bf, int Nn, int Fin, int C, int pbase,
                         const int* esrc, const int* edst, int E,
                         float* xl, float* xr, float* acc,
                         float* esc, float* ewt, unsigned* emx, float* den,
                         int concat, const float* res,
                         float* out, u16t* outbf, int transbf) {
        cvtT(F(pbase + 0), wbufL, Fin, NHEADS * C);
        cvtT(F(pbase + 2), wbufR, Fin, NHEADS * C);
        const float* bl = F(pbase + 1); const float* br = F(pbase + 3);
        const float* att = F(pbase + 4); const float* bias = F(pbase + 5);
        int HC = NHEADS * C;
        gemm(A_bf, wbufL, bl, xl, nullptr, Nn, HC, Fin, HC, 0, 2);
        gemm(A_bf, wbufR, br, xr, nullptr, Nn, HC, Fin, HC, 0, 2);
        k_fillu<<<cdiv(Nn * NHEADS, 256), 256, 0, stream>>>(emx, 0x00800000u, Nn * NHEADS); // enc(-FLT_MAX)
        k_fillf<<<cdiv(Nn * NHEADS, 256), 256, 0, stream>>>(den, 0.f, Nn * NHEADS);
        k_fillf<<<cdiv(Nn * HC, 256), 256, 0, stream>>>(acc, 0.f, Nn * HC);
        int eb = cdiv(E + Nn, 8); if (eb > 2048) eb = 2048;
        k_edge_scores<<<eb, 256, 0, stream>>>(xl, xr, att, esrc, edst, E, Nn, C, esc, emx);
        k_edge_soft<<<cdiv((E + Nn) * NHEADS, 256), 256, 0, stream>>>(esc, edst, E, Nn, emx, ewt, den);
        k_edge_scatter<<<eb, 256, 0, stream>>>(xl, esrc, edst, E, Nn, C, ewt, den, acc);
        k_gat_post<<<cdiv(Nn, 8), 256, 0, stream>>>(acc, bias, res, out, outbf, transbf,
                                                    Nn, NHEADS, C, concat, 0);
    };

    // ===== forward =====
    k_xg<<<cdiv(BB * GG, 256), 256, 0, stream>>>(x_genes, cl, ev, bt, alpha, beta, gamma,
                                                 xg_bf, xgT_bf);
    k_cvt<<<cdiv(BB * NCC, 256), 256, 0, stream>>>(clinical, clin_bf, BB * NCC);

    // gene-graph GAT stack over [G, ...]
    gat_layer(xgT_bf, GG, 128, 128, 24, gsrc, gdst, EGG, gxl, gxr, gacc, gesc, gewt, gemx, gden,
              1, nullptr, nullptr, gh1_bf, 0);
    gat_layer(gh1_bf, GG, 256, 128, 30, gsrc, gdst, EGG, gxl, gxr, gacc, gesc, gewt, gemx, gden,
              0, nullptr, gh2, gh2_bf, 0);
    gat_layer(gh2_bf, GG, 128, 128, 36, gsrc, gdst, EGG, gxl, gxr, gacc, gesc, gewt, gemx, gden,
              0, gh2, nullptr, gh3T_bf, 1);   // bf16 mirror stored transposed: [128][G]

    // xge = h^T @ fc_gene -> comb[:, 0:512] (bf16)
    cvtT(fgW, wbufL, GG, HDD);
    gemm(gh3T_bf, wbufL, fgb, nullptr, comb_bf, BB, HDD, GG, 1056, 0, 2);
    // q = xg @ fc_g2p (bf16)
    cvtT(g2pW, wbufR, GG, HDD);
    gemm(xg_bf, wbufR, g2pb, nullptr, pq_bf, BB, HDD, GG, HDD, 0, 2);

    // patient-graph GAT stack over [B, ...]
    gat_layer(pq_bf,  BB, 512, 512, 42, psrc, pdst, EPP, pxl, pxr, pacc, pesc, pewt, pemx, pden,
              1, nullptr, nullptr, ph1_bf, 0);
    gat_layer(ph1_bf, BB, 1024, 512, 48, psrc, pdst, EPP, pxl, pxr, pacc, pesc, pewt, pemx, pden,
              0, nullptr, ph2, ph2_bf, 0);
    gat_layer(ph2_bf, BB, 512, 512, 54, psrc, pdst, EPP, pxl, pxr, pacc, pesc, pewt, pemx, pden,
              0, ph2, nullptr, ph3_bf, 0);

    // xpe -> comb[:, 512:1024]; clin -> comb[:, 1024:1056]
    cvtT(fpW, wbufL, HDD, HDD);
    gemm(ph3_bf, wbufL, fpb, nullptr, comb_bf, BB, HDD, HDD, 1056, 512, 2);
    cvtT(clinW, wbufR, NCC, NCC);
    gemm(clin_bf, wbufR, clinB, nullptr, comb_bf, BB, NCC, NCC, 1056, 1024, 2);
    // (MHA branch is multiplied by exactly 0.0 downstream -> eliminated)

    // risk head
    cvtT(r1W, wbufL, 1056, 512);
    gemm(comb_bf, wbufL, r1b, nullptr, r1_bf, BB, 512, 1056, 512, 0, 1);
    cvtT(r2W, wbufR, 512, 64);
    gemm(r1_bf, wbufR, r2b, r2, nullptr, BB, 64, 512, 64, 0, 1);
    k_dot<<<16, 256, 0, stream>>>(r2, r3W, r3b, (float*)d_out + BB, BB, 64);

    // out head
    cvtT(fc1W, wbufL, 1056, 256);
    gemm(comb_bf, wbufL, fc1b, t1, nullptr, BB, 256, 1056, 256, 0, 2);
    k_gat_post<<<16, 256, 0, stream>>>(t1, nullptr, nullptr, nullptr, t1_bf, 0,
                                       BB, 1, 256, 1, 1);   // relu(ln(x)) -> bf16
    cvtT(fc2W, wbufR, 256, 32);
    gemm(t1_bf, wbufR, fc2b, t2, nullptr, BB, 32, 256, 32, 0, 2);
    k_gat_post<<<16, 256, 0, stream>>>(t2, nullptr, nullptr, t2, nullptr, 0,
                                       BB, 1, 32, 1, 1);
    k_dot<<<16, 256, 0, stream>>>(t2, fc3W, fc3b, (float*)d_out, BB, 32);
}
